// MLPLatentWorldModel_4389456576605
// MI455X (gfx1250) — matrix operand order repaired and checked by reading, not compiled, not run on hardware
//
#include <hip/hip_runtime.h>

// ---------------- shapes ----------------
#define TT    64
#define BB    256
#define OBSD  64
#define ACTD  16
#define HIDD  1024
#define REPRD 512
#define GH    1024
#define H3    3072
#define H4    4096
#define KC    1568   // 512+16+1024 padded to mult of 32
#define KO    1056   // 16+1024 padded to mult of 32
#define MENC  (TT*BB)

// async global->LDS staging if the gfx1250 builtins are available
#if defined(__gfx1250__) && __has_builtin(__builtin_amdgcn_global_load_async_to_lds_b64) && __has_builtin(__builtin_amdgcn_s_wait_asynccnt)
#define ASYNC_STAGE 1
#else
#define ASYNC_STAGE 0
#endif

typedef __attribute__((ext_vector_type(16))) __bf16 v16bf;
typedef __attribute__((ext_vector_type(8)))  float  v8f;

#if ASYNC_STAGE
typedef int v2i_vs __attribute__((vector_size(8)));             // matches builtin proto V2i
typedef __attribute__((address_space(1))) v2i_vs* as1_v2i;      // global v2i*
typedef __attribute__((address_space(3))) v2i_vs* as3_v2i;      // LDS v2i*
#endif

union BFrag { v16bf v; unsigned u[8]; };

__device__ __forceinline__ unsigned short f2bf(float f) {
  unsigned u = __float_as_uint(f);
  u += 0x7fffu + ((u >> 16) & 1u);       // round-to-nearest-even
  return (unsigned short)(u >> 16);
}
__device__ __forceinline__ float sigm(float x){
  // fast sigmoid: v_exp_f32 + v_rcp_f32 (no IEEE divide)
  return __builtin_amdgcn_rcpf(1.f + __expf(-x));
}
__device__ __forceinline__ float tanh_fast(float x){
  return 2.f * sigm(2.f * x) - 1.f;
}

// ---------------- shared WMMA main loop ----------------
// 128x128 block tile, BK=32, 8 waves (2x4), each wave 64x32 (4x2 WMMA tiles).
// Double-buffered LDS; staging via ASYNCcnt-tracked global_load_async_to_lds
// when available, else register-staged copies. 1 barrier / K-step.
__device__ __forceinline__ void mm_mainloop(
    const unsigned short* __restrict__ A, int lda,
    const unsigned short* __restrict__ Bt, int ldb,
    long bm, long bn, int K, int tid,
    unsigned short* sA0, unsigned short* sB0, v8f (&acc)[4][2])
{
  const int lane = tid & 31;
  const int rowW = ((tid >> 5) >> 2) * 64;
  const int colW = ((tid >> 5) & 3) * 32;
  const int m    = lane & 15;
  const int hlf  = lane >> 4;

  auto do_tile = [&](int buf){
    const unsigned short* pa = sA0 + buf*(128*36);
    const unsigned short* pb = sB0 + buf*(128*36);
    BFrag af[4], bfr[2];
#pragma unroll
    for (int mt=0; mt<4; mt++){
      int r = rowW + mt*16 + m;
#pragma unroll
      for (int v=0; v<8; v++){            // ISA 16-bit A 16x32 lane layout
        int k = (v < 4) ? (hlf*8 + v*2) : (16 + hlf*8 + (v-4)*2);
        af[mt].u[v] = *(const unsigned*)(&pa[r*36 + k]);
      }
    }
#pragma unroll
    for (int nt=0; nt<2; nt++){           // B 32x16: lane=col, K-pairs in VGPRs
      int n  = colW + nt*16 + m;
#pragma unroll
      for (int v=0; v<8; v++){
        int kb = (v < 4) ? (hlf*8 + v*2) : (16 + hlf*8 + (v-4)*2);
        bfr[nt].u[v] = *(const unsigned*)(&pb[n*36 + kb]);
      }
    }
#pragma unroll
    for (int mt=0; mt<4; mt++)
#pragma unroll
      for (int nt=0; nt<2; nt++)
        acc[mt][nt] = __builtin_amdgcn_wmma_f32_16x16x32_bf16(
            false, af[mt].v, false, bfr[nt].v, (short)0, acc[mt][nt], false, false);
  };

#if ASYNC_STAGE
  // b64 chunks: 128 rows x 8 chunks(4 elem) per matrix; 4 A + 4 B asyncs/thread
  auto issue = [&](int buf, int kt){
    unsigned short* pa = sA0 + buf*(128*36);
    unsigned short* pb = sB0 + buf*(128*36);
#pragma unroll
    for (int i=0;i<4;i++){
      int q = tid + i*256;
      int row = q >> 3, ch = (q & 7) * 4;
      const unsigned short* ga = A  + (size_t)(bm + row)*(size_t)lda + kt + ch;
      const unsigned short* gb = Bt + (size_t)(bn + row)*(size_t)ldb + kt + ch;
      __builtin_amdgcn_global_load_async_to_lds_b64(
          (as1_v2i)(void*)ga, (as3_v2i)(void*)(pa + row*36 + ch), 0, 0);
      __builtin_amdgcn_global_load_async_to_lds_b64(
          (as1_v2i)(void*)gb, (as3_v2i)(void*)(pb + row*36 + ch), 0, 0);
    }
  };

  issue(0, 0);
  int cur = 0;
  for (int kt = 0; kt < K; kt += 32) {
    const bool more = (kt + 32) < K;
    if (kt + 64 < K) {                   // warm caches for the tile after next
      int pr = ((tid >> 4) * 8) + (tid & 7);
      __builtin_prefetch(A  + (size_t)(bm + pr)*(size_t)lda + kt + 64, 0, 1);
      __builtin_prefetch(Bt + (size_t)(bn + pr)*(size_t)ldb + kt + 64, 0, 1);
    }
    __builtin_amdgcn_s_wait_asynccnt(0); // my fills for buf[cur] landed
    __syncthreads();                     // everyone's landed; prev compute done
    if (more) issue(cur ^ 1, kt + 32);   // async engine fills under the WMMAs
    do_tile(cur);
    cur ^= 1;
  }
#else
  const int lc2 = (tid & 15) * 2;
  const int lr0 = (tid >> 4) * 8;
  unsigned ra[8], rb[8];
  auto gload = [&](int kt){
#pragma unroll
    for (int i=0;i<8;i++){
      int r = lr0 + i;
      ra[i] = *(const unsigned*)(A  + (size_t)(bm + r)*(size_t)lda + kt + lc2);
      rb[i] = *(const unsigned*)(Bt + (size_t)(bn + r)*(size_t)ldb + kt + lc2);
    }
  };
  auto lstore = [&](int buf){
    unsigned short* pa = sA0 + buf*(128*36);
    unsigned short* pb = sB0 + buf*(128*36);
#pragma unroll
    for (int i=0;i<8;i++){
      int r = lr0 + i;
      *(unsigned*)(&pa[r*36 + lc2]) = ra[i];
      *(unsigned*)(&pb[r*36 + lc2]) = rb[i];
    }
  };
  gload(0);
  lstore(0);
  int cur = 0;
  for (int kt = 0; kt < K; kt += 32) {
    const bool more = (kt + 32) < K;
    if (more) gload(kt + 32);
    if (kt + 64 < K) {
      int pr = lr0 + (tid & 7);
      __builtin_prefetch(A  + (size_t)(bm + pr)*(size_t)lda + kt + 64, 0, 1);
      __builtin_prefetch(Bt + (size_t)(bn + pr)*(size_t)ldb + kt + 64, 0, 1);
    }
    __syncthreads();
    do_tile(cur);
    if (more) lstore(cur ^ 1);
    cur ^= 1;
  }
#endif
}

// ---------------- generic GEMM: C = act(A@Bt^T + bias) ----------------
template<int ACT, int WF, int WB>
__global__ __launch_bounds__(256) void gemm_bf16(
    const unsigned short* __restrict__ A, int lda,
    const unsigned short* __restrict__ Bt, int ldb,
    const float* __restrict__ bias,
    float* __restrict__ outF, unsigned short* __restrict__ outB,
    int N, int K)
{
  __shared__ __align__(16) unsigned char smem[36864];
  unsigned short* sA0 = (unsigned short*)smem;
  unsigned short* sB0 = (unsigned short*)(smem + 18432);

  const int tid  = threadIdx.x;
  const int lane = tid & 31;
  const int rowW = ((tid >> 5) >> 2) * 64;
  const int colW = ((tid >> 5) & 3) * 32;
  const int m    = lane & 15;
  const int hlf  = lane >> 4;
  const long bm = (long)blockIdx.y * 128;
  const long bn = (long)blockIdx.x * 128;

  v8f acc[4][2];
  v8f z8 = {0.f,0.f,0.f,0.f,0.f,0.f,0.f,0.f};
#pragma unroll
  for (int i=0;i<4;i++)
#pragma unroll
    for (int j=0;j<2;j++) acc[i][j]=z8;

  mm_mainloop(A, lda, Bt, ldb, bm, bn, K, tid, sA0, sB0, acc);

#pragma unroll
  for (int mt=0; mt<4; mt++)
#pragma unroll
    for (int nt=0; nt<2; nt++){
      long rb2 = bm + rowW + mt*16 + hlf*8;     // C layout: lanes16-31 -> rows+8
      long col = bn + colW + nt*16 + m;
      float bv = bias ? bias[col] : 0.f;
#pragma unroll
      for (int vg=0; vg<8; vg++){
        float x = acc[mt][nt][vg] + bv;
        if (ACT) x = x * sigm(x);               // swish
        size_t idx = (size_t)(rb2 + vg) * (size_t)N + (size_t)col;
        if (WF) outF[idx] = x;
        if (WB) outB[idx] = f2bf(x);
      }
    }
}

// ---------------- fused GRU-step GEMM ----------------
// Weight is gate-interleaved: output col n = 4*j + g, g in {z, r, a_x, a_h}.
// N fixed = 4096. Epilogue stages gates in LDS (reusing stage bufs), computes
// h' = (1-z)h + z*tanh(a_x + r*a_h), writes f32 h chain, bf16 state row, and
// bf16 h into the NEXT step's A panel.
__global__ __launch_bounds__(256) void gemm_gru(
    const unsigned short* __restrict__ A, int lda,
    const unsigned short* __restrict__ Bt, int ldb,
    const float* __restrict__ b3,
    float* __restrict__ hf, const float* __restrict__ hprev,
    unsigned short* __restrict__ stOut, unsigned short* __restrict__ nextA,
    int K)
{
  __shared__ __align__(16) unsigned char smem[36864];
  unsigned short* sA0 = (unsigned short*)smem;
  unsigned short* sB0 = (unsigned short*)(smem + 18432);
  float (*gS)[132] = (float (*)[132])smem;     // 64x132 f32 = 33792B, reused post-loop

  const int tid  = threadIdx.x;
  const int lane = tid & 31;
  const int rowW = ((tid >> 5) >> 2) * 64;
  const int colW = ((tid >> 5) & 3) * 32;
  const int m    = lane & 15;
  const int hlf  = lane >> 4;
  const long bm = (long)blockIdx.y * 128;
  const long bn = (long)blockIdx.x * 128;

  v8f acc[4][2];
  v8f z8 = {0.f,0.f,0.f,0.f,0.f,0.f,0.f,0.f};
#pragma unroll
  for (int i=0;i<4;i++)
#pragma unroll
    for (int j=0;j<2;j++) acc[i][j]=z8;

  mm_mainloop(A, lda, Bt, ldb, bm, bn, K, tid, sA0, sB0, acc);

  const int jb = (int)(bn >> 2);               // 32 consecutive j's per block
#pragma unroll
  for (int half = 0; half < 2; half++){
    __syncthreads();                           // prior LDS users done
    if ((rowW >> 6) == half) {                 // 4 waves own this 64-row half
#pragma unroll
      for (int mt=0; mt<4; mt++)
#pragma unroll
        for (int nt=0; nt<2; nt++){
          int rl = mt*16 + hlf*8;
          int cl = colW + nt*16 + m;
#pragma unroll
          for (int vg=0; vg<8; vg++)
            gS[rl+vg][cl] = acc[mt][nt][vg];
        }
    }
    __syncthreads();
#pragma unroll
    for (int e=0; e<8; e++){                   // 64*32 elems / 256 threads
      int idx = tid + e*256;
      int rl  = idx >> 5;                      // 0..63
      int jl  = idx & 31;
      int j   = jb + jl;
      long b  = bm + half*64 + rl;
      float z = sigm(gS[rl][4*jl+0] + b3[j]);
      float r = sigm(gS[rl][4*jl+1] + b3[GH + j]);
      float a = tanh_fast(gS[rl][4*jl+2] + r * gS[rl][4*jl+3] + b3[2*GH + j]);
      float hp = hprev ? hprev[(size_t)b*GH + j] : 0.f;
      float hn = (1.f - z)*hp + z*a;
      hf[(size_t)b*GH + j]    = hn;
      stOut[(size_t)b*GH + j] = f2bf(hn);
      if (nextA) nextA[(size_t)b*KO + ACTD + j] = f2bf(hn);
    }
  }
}

// ---------------- helper kernels ----------------
__global__ void k_f32_to_bf16(unsigned short* __restrict__ d, const float* __restrict__ s, long n){
  long i = (long)blockIdx.x*blockDim.x + threadIdx.x;
  if (i < n) d[i] = f2bf(s[i]);
}
// dst[N,K] = bf16(src[K,N])
__global__ void k_transpose_bf16(unsigned short* __restrict__ d, const float* __restrict__ s, int K, int N){
  long i = (long)blockIdx.x*blockDim.x + threadIdx.x;
  if (i >= (long)K*N) return;
  int k = (int)(i % K); int n = (int)(i / K);
  d[(size_t)n*K + k] = f2bf(s[(size_t)k*N + n]);
}
// gate-interleaved transposed GRU weight: out col n = 4*j + g
// g=0: z (wi|wh col j), g=1: r (wi|wh col H+j), g=2: a_x (wi col 2H+j), g=3: a_h (wh col 2H+j)
__global__ void k_build_gru_wt(unsigned short* __restrict__ d, const float* __restrict__ wi,
                               const float* __restrict__ wh, int Kx, int Kh, int Kpad, int H){
  long i = (long)blockIdx.x*blockDim.x + threadIdx.x;
  long total = (long)4*H*Kpad;
  if (i >= total) return;
  int k = (int)(i % Kpad); int n = (int)(i / Kpad);
  int j = n >> 2, g = n & 3;
  float v = 0.f;
  if (k < Kx + Kh) {
    if (g == 0)      v = (k < Kx) ? wi[(size_t)k*3*H + j]     : wh[(size_t)(k-Kx)*3*H + j];
    else if (g == 1) v = (k < Kx) ? wi[(size_t)k*3*H + H + j] : wh[(size_t)(k-Kx)*3*H + H + j];
    else if (g == 2) { if (k < Kx)  v = wi[(size_t)k*3*H + 2*H + j]; }
    else             { if (k >= Kx) v = wh[(size_t)(k-Kx)*3*H + 2*H + j]; }
  }
  d[(size_t)n*Kpad + k] = f2bf(v);
}
// closed-step A panel: [emb0 | act0 | zeros(h0)+pad]
__global__ void k_pack_closed(unsigned short* __restrict__ A, const float* __restrict__ emb0,
                              const float* __restrict__ act0){
  long i = (long)blockIdx.x*blockDim.x + threadIdx.x;
  if (i >= (long)BB*KC) return;
  int c = (int)(i % KC); int b = (int)(i / KC);
  float v = 0.f;
  if (c < REPRD)            v = emb0[(size_t)b*REPRD + c];
  else if (c < REPRD+ACTD)  v = act0[(size_t)b*ACTD + (c - REPRD)];
  A[i] = f2bf(v);
}
// open-step A panels for all steps: [action_s | (h filled later) | pad]
__global__ void k_pack_open(unsigned short* __restrict__ A, const float* __restrict__ actions){
  long i = (long)blockIdx.x*blockDim.x + threadIdx.x;
  long total = (long)TT*BB*KO;
  if (i >= total) return;
  int c = (int)(i % KO);
  long sb = i / KO;
  int b = (int)(sb % BB);
  int s = (int)(sb / BB);
  float v = 0.f;
  if (c < ACTD) v = actions[((size_t)s*BB + b)*ACTD + c];
  A[i] = f2bf(v);
}

// ---------------- host ----------------
static inline long cdivl(long a, long b){ return (a + b - 1) / b; }

extern "C" void kernel_launch(void* const* d_in, const int* in_sizes, int n_in,
                              void* d_out, int out_size, void* d_ws, size_t ws_size,
                              hipStream_t stream) {
  (void)in_sizes; (void)n_in; (void)out_size; (void)ws_size;
  // jax pytree (sorted-key) flat order after obs, actions:
  const float* obs       = (const float*)d_in[0];
  const float* actions   = (const float*)d_in[1];
  const float* closed_b  = (const float*)d_in[2];
  const float* closed_wh = (const float*)d_in[3];
  const float* closed_wi = (const float*)d_in[4];
  const float* enc_b[5]; const float* enc_w[5];
  for (int i=0;i<5;i++){ enc_b[i]=(const float*)d_in[5+2*i]; enc_w[i]=(const float*)d_in[6+2*i]; }
  const float* open_b    = (const float*)d_in[15];
  const float* open_wh   = (const float*)d_in[16];
  const float* open_wi   = (const float*)d_in[17];
  const float* pred0_b   = (const float*)d_in[18];
  const float* pred0_w   = (const float*)d_in[19];
  const float* pred1_b   = (const float*)d_in[20];
  const float* pred1_w   = (const float*)d_in[21];

  float* out_pred = (float*)d_out;                              // (64,256,512)
  float* out_emb  = (float*)d_out + (size_t)MENC*REPRD;         // (64,256,512)

  // workspace carve-up
  char* p = (char*)d_ws;
  auto abf = [&](size_t n){ unsigned short* r=(unsigned short*)p; p += ((n*2+255)/256)*256; return r; };
  auto af32= [&](size_t n){ float* r=(float*)p;                  p += ((n*4+255)/256)*256; return r; };
  unsigned short* WtE0 = abf((size_t)HIDD*OBSD);
  unsigned short* WtE1 = abf((size_t)HIDD*HIDD);
  unsigned short* WtE2 = abf((size_t)HIDD*HIDD);
  unsigned short* WtE3 = abf((size_t)HIDD*HIDD);
  unsigned short* WtE4 = abf((size_t)REPRD*HIDD);
  unsigned short* WtP0 = abf((size_t)HIDD*GH);
  unsigned short* WtP1 = abf((size_t)REPRD*HIDD);
  unsigned short* WtC  = abf((size_t)H4*KC);
  unsigned short* WtO  = abf((size_t)H4*KO);
  unsigned short* obsB = abf((size_t)MENC*OBSD);
  unsigned short* bufX = abf((size_t)MENC*HIDD);
  unsigned short* bufY = abf((size_t)MENC*HIDD);
  unsigned short* states = abf((size_t)TT*BB*GH);
  unsigned short* Aopen  = abf((size_t)TT*BB*KO);
  unsigned short* Aclosed= abf((size_t)BB*KC);
  float* hf32 = af32((size_t)BB*GH);

  const int TB = 256;
  // ---- weight prep (bf16, transposed N x K) ----
  k_transpose_bf16<<<cdivl((long)OBSD*HIDD,TB),TB,0,stream>>>(WtE0, enc_w[0], OBSD, HIDD);
  k_transpose_bf16<<<cdivl((long)HIDD*HIDD,TB),TB,0,stream>>>(WtE1, enc_w[1], HIDD, HIDD);
  k_transpose_bf16<<<cdivl((long)HIDD*HIDD,TB),TB,0,stream>>>(WtE2, enc_w[2], HIDD, HIDD);
  k_transpose_bf16<<<cdivl((long)HIDD*HIDD,TB),TB,0,stream>>>(WtE3, enc_w[3], HIDD, HIDD);
  k_transpose_bf16<<<cdivl((long)HIDD*REPRD,TB),TB,0,stream>>>(WtE4, enc_w[4], HIDD, REPRD);
  k_transpose_bf16<<<cdivl((long)GH*HIDD,TB),TB,0,stream>>>(WtP0, pred0_w, GH, HIDD);
  k_transpose_bf16<<<cdivl((long)HIDD*REPRD,TB),TB,0,stream>>>(WtP1, pred1_w, HIDD, REPRD);
  k_build_gru_wt<<<cdivl((long)H4*KC,TB),TB,0,stream>>>(WtC, closed_wi, closed_wh, REPRD+ACTD, GH, KC, GH);
  k_build_gru_wt<<<cdivl((long)H4*KO,TB),TB,0,stream>>>(WtO, open_wi,   open_wh,   ACTD,       GH, KO, GH);
  k_pack_open<<<cdivl((long)TT*BB*KO,TB),TB,0,stream>>>(Aopen, actions);

  // ---- encoder MLP (M=16384) ----
  k_f32_to_bf16<<<cdivl((long)MENC*OBSD,TB),TB,0,stream>>>(obsB, obs, (long)MENC*OBSD);
  gemm_bf16<1,0,1><<<dim3(HIDD/128, MENC/128),TB,0,stream>>>(obsB, OBSD, WtE0, OBSD, enc_b[0], nullptr, bufX, HIDD, OBSD);
  gemm_bf16<1,0,1><<<dim3(HIDD/128, MENC/128),TB,0,stream>>>(bufX, HIDD, WtE1, HIDD, enc_b[1], nullptr, bufY, HIDD, HIDD);
  gemm_bf16<1,0,1><<<dim3(HIDD/128, MENC/128),TB,0,stream>>>(bufY, HIDD, WtE2, HIDD, enc_b[2], nullptr, bufX, HIDD, HIDD);
  gemm_bf16<1,0,1><<<dim3(HIDD/128, MENC/128),TB,0,stream>>>(bufX, HIDD, WtE3, HIDD, enc_b[3], nullptr, bufY, HIDD, HIDD);
  gemm_bf16<0,1,0><<<dim3(REPRD/128,MENC/128),TB,0,stream>>>(bufY, HIDD, WtE4, HIDD, enc_b[4], out_emb, nullptr, REPRD, HIDD);

  // ---- closed GRU step (fused gates; -> states[0], seeds Aopen[1]) ----
  k_pack_closed<<<cdivl((long)BB*KC,TB),TB,0,stream>>>(Aclosed, out_emb, actions);
  gemm_gru<<<dim3(H4/128, BB/128),TB,0,stream>>>(Aclosed, KC, WtC, KC, closed_b,
      hf32, nullptr, states, Aopen + (size_t)1*BB*KO, KC);

  // ---- open GRU scan, steps 1..63: ONE fused kernel per step ----
  for (int s = 1; s < TT; ++s) {
    gemm_gru<<<dim3(H4/128, BB/128),TB,0,stream>>>(Aopen + (size_t)s*BB*KO, KO, WtO, KO, open_b,
        hf32, hf32, states + (size_t)s*BB*GH,
        (s+1 < TT) ? (Aopen + (size_t)(s+1)*BB*KO) : nullptr, KO);
  }

  // ---- predictor MLP over all 64*256 states ----
  gemm_bf16<1,0,1><<<dim3(HIDD/128, MENC/128),TB,0,stream>>>(states, GH, WtP0, GH, pred0_b, nullptr, bufX, HIDD, GH);
  gemm_bf16<0,1,0><<<dim3(REPRD/128,MENC/128),TB,0,stream>>>(bufX, HIDD, WtP1, HIDD, pred1_b, out_pred, nullptr, REPRD, HIDD);
}
